// SchNETInteractionModule_5437428597389
// MI455X (gfx1250) — compile-verified, matrix operand order checked by reading, not run on hardware
//
#include <hip/hip_runtime.h>

typedef __attribute__((ext_vector_type(16))) _Float16 v16h;
typedef __attribute__((ext_vector_type(8)))  _Float16 v8h;
typedef __attribute__((ext_vector_type(8)))  float    v8f;

#define WMMA_F16(a, b, c) \
  __builtin_amdgcn_wmma_f32_16x16x32_f16(false, (a), false, (b), (short)0, (c), false, false)

#define LN2F 0.69314718055994531f

// shifted softplus: softplus(x) - ln(2), via native v_exp_f32 / v_log_f32.
// t = e^{-|x|} in (0,1], so log(1+t) is well-conditioned.
__device__ __forceinline__ float sspf(float x) {
  float t = __expf(-fabsf(x));
  return fmaxf(x, 0.0f) + __logf(1.0f + t) - LN2F;
}

// Convert 16 f32 (32B chunks at p[0..7] and p[16..23]) into a WMMA f16 fragment.
// Matches the 16-bit A/B per-lane layout: chunk0 = K+0..7, chunk1 = K+16..23.
__device__ __forceinline__ v16h cvt_frag_f32(const float* __restrict__ p) {
  float4 c0 = *(const float4*)(p + 0);
  float4 c1 = *(const float4*)(p + 4);
  float4 c2 = *(const float4*)(p + 16);
  float4 c3 = *(const float4*)(p + 20);
  v16h f;
  f[0]  = (_Float16)c0.x; f[1]  = (_Float16)c0.y; f[2]  = (_Float16)c0.z; f[3]  = (_Float16)c0.w;
  f[4]  = (_Float16)c1.x; f[5]  = (_Float16)c1.y; f[6]  = (_Float16)c1.z; f[7]  = (_Float16)c1.w;
  f[8]  = (_Float16)c2.x; f[9]  = (_Float16)c2.y; f[10] = (_Float16)c2.z; f[11] = (_Float16)c2.w;
  f[12] = (_Float16)c3.x; f[13] = (_Float16)c3.y; f[14] = (_Float16)c3.z; f[15] = (_Float16)c3.w;
  return f;
}

// Fragment from row-major padded f16 LDS tile (chunks at +0 and +16 halves).
__device__ __forceinline__ v16h lds_frag(const _Float16* __restrict__ p) {
  v8h lo = *(const v8h*)(p);        // ds_load_b128
  v8h hi = *(const v8h*)(p + 16);   // ds_load_b128
  v16h f;
  #pragma unroll
  for (int i = 0; i < 8; ++i) { f[i] = lo[i]; f[8 + i] = hi[i]; }
  return f;
}

// Build a fragment-swizzled f16 copy of W[K x 128] (f32, row-major) in LDS.
// dst frag id g = (kk*8 + t)*32 + lane; each lane's 16 halves are contiguous,
// so a B-fragment at use time is a single 32B v16h load (2x ds_load_b128).
__device__ __forceinline__ void build_frags(_Float16* __restrict__ dst,
                                            const float* __restrict__ W,
                                            int nkk, int tid) {
  const int nfrag = nkk * 8 * 32;
  for (int g = tid; g < nfrag; g += 128) {
    const int lane = g & 31;
    const int t    = (g >> 5) & 7;
    const int kk   = g >> 8;
    const int n    = t * 16 + (lane & 15);
    const int kc   = kk * 32 + ((lane >> 4) << 3);
    const float* w0 = W + kc * 128 + n;
    v16h f;
    #pragma unroll
    for (int i = 0; i < 8; ++i) {
      f[i]     = (_Float16)w0[i * 128];
      f[8 + i] = (_Float16)w0[(16 + i) * 128];
    }
    *(v16h*)(dst + g * 16) = f;     // 2x ds_store_b128
  }
}

__device__ __forceinline__ v16h bfrag(const _Float16* __restrict__ sWs,
                                      int kk, int t, int lane) {
  return *(const v16h*)(sWs + ((kk * 8 + t) * 32 + lane) * 16);
}

// ---------------------------------------------------------------------------
// Kernel A: h[N,128] = x[N,128] @ W_in[128,128]
// ---------------------------------------------------------------------------
__global__ void __launch_bounds__(128) k_in_proj(const float* __restrict__ x,
                                                 const float* __restrict__ Win,
                                                 float* __restrict__ h, int N) {
  __shared__ _Float16 sWs[16384];               // 32 KB, fragment-swizzled
  const int tid = threadIdx.x;
  build_frags(sWs, Win, 4, tid);
  __syncthreads();

  const int lane = tid & 31;
  const int wave = tid >> 5;
  const int rowBase = (blockIdx.x * 4 + wave) * 16;
  const int m  = lane & 15;
  const int kb = (lane >> 4) << 3;
  const int mg = kb;
  const bool full = (rowBase + 16 <= N);

  int row = rowBase + m; if (row >= N) row = N - 1;
  const float* xr = x + (long long)row * 128;

  v16h a[4];
  #pragma unroll
  for (int kk = 0; kk < 4; ++kk) a[kk] = cvt_frag_f32(xr + kk * 32 + kb);

  #pragma unroll
  for (int t = 0; t < 8; ++t) {
    v8f c = {};
    #pragma unroll
    for (int kk = 0; kk < 4; ++kk) c = WMMA_F16(a[kk], bfrag(sWs, kk, t, lane), c);
    const int n = t * 16 + m;
    float* hp = h + (long long)(rowBase + mg) * 128 + n;
    if (full) {
      #pragma unroll
      for (int rr = 0; rr < 8; ++rr) hp[rr * 128] = c[rr];
    } else {
      #pragma unroll
      for (int rr = 0; rr < 8; ++rr)
        if (rowBase + mg + rr < N) hp[rr * 128] = c[rr];
    }
  }
}

// ---------------------------------------------------------------------------
// Kernel B: fused per-pair filter network + gather + scatter-add.
//   W1 = ssp(f_ij @ Wf1 + bf1); W2 = (W1 @ Wf2 + bf2) * cutoff
//   acc[idx_i] += h[idx_j] * W2     (f32 atomics, L2-resident)
// ---------------------------------------------------------------------------
__global__ void __launch_bounds__(128) k_pairs(const float* __restrict__ fij,
                                               const float* __restrict__ fcut,
                                               const int*   __restrict__ pl,
                                               const float* __restrict__ Wf1,
                                               const float* __restrict__ bf1,
                                               const float* __restrict__ Wf2,
                                               const float* __restrict__ bf2,
                                               const float* __restrict__ h,
                                               float* __restrict__ acc,
                                               int P) {
  __shared__ _Float16 sW1s[4096];               //  8 KB, Wf1 frag-swizzled
  __shared__ _Float16 sW2s[16384];              // 32 KB, Wf2 frag-swizzled
  __shared__ _Float16 sA[4][16 * 136];          // per-wave C->A transpose tile
  const int tid = threadIdx.x;
  build_frags(sW1s, Wf1, 1, tid);
  build_frags(sW2s, Wf2, 4, tid);
  __syncthreads();

  const int lane = tid & 31;
  const int wave = tid >> 5;
  const long long pBase = ((long long)blockIdx.x * 4 + wave) * 16;
  const int m  = lane & 15;
  const int kb = (lane >> 4) << 3;
  const int mg = kb;
  const bool full = (pBase + 16 <= (long long)P);

  long long prow = pBase + m; if (prow >= P) prow = P - 1;
  v16h a1 = cvt_frag_f32(fij + prow * 32 + kb);

  _Float16* myA = &sA[wave][0];
  // GEMM1 (K=32) + bias + SSP -> f16 LDS tile
  #pragma unroll
  for (int t = 0; t < 8; ++t) {
    v8f c = {};
    c = WMMA_F16(a1, bfrag(sW1s, 0, t, lane), c);
    const int n = t * 16 + m;
    const float bias = bf1[n];
    _Float16* mp = myA + mg * 136 + n;
    #pragma unroll
    for (int rr = 0; rr < 8; ++rr) mp[rr * 136] = (_Float16)sspf(c[rr] + bias);
  }
  __syncthreads();

  // GEMM2 (K=128)
  v8f c2[8];
  v8f zero = {};
  #pragma unroll
  for (int t = 0; t < 8; ++t) c2[t] = zero;
  #pragma unroll
  for (int kk = 0; kk < 4; ++kk) {
    v16h a2 = lds_frag(myA + m * 136 + kk * 32 + kb);
    #pragma unroll
    for (int t = 0; t < 8; ++t) c2[t] = WMMA_F16(a2, bfrag(sW2s, kk, t, lane), c2[t]);
  }

  float b2[8];
  #pragma unroll
  for (int t = 0; t < 8; ++t) b2[t] = bf2[t * 16 + m];

  // epilogue: cutoff scale, gather h[idx_j] (L2-hot), atomic scatter to acc[idx_i]
  #pragma unroll
  for (int rr = 0; rr < 8; ++rr) {
    const long long p = pBase + mg + rr;
    if (full || p < P) {
      const int ii = pl[p];                     // pairlist row 0: idx_i
      const int jj = pl[(long long)P + p];      // pairlist row 1: idx_j
      const float cut = fcut[p];
      const float* hrow = h + (long long)jj * 128 + m;
      float* arow = acc + (long long)ii * 128 + m;
      #pragma unroll
      for (int t = 0; t < 8; ++t) {
        const float w = (c2[t][rr] + b2[t]) * cut;
        atomicAdd(&arow[t * 16], hrow[t * 16] * w);   // global_atomic_add_f32
      }
    }
  }
}

// ---------------------------------------------------------------------------
// Kernel C: out = ssp(acc @ Wo1 + bo1) @ Wo2 + bo2
// One 32KB swizzle buffer, phase-shared: Wo1 frags -> GEMM1 -> Wo2 frags -> GEMM2
// ---------------------------------------------------------------------------
__global__ void __launch_bounds__(128) k_out_mlp(const float* __restrict__ acc,
                                                 const float* __restrict__ Wo1,
                                                 const float* __restrict__ bo1,
                                                 const float* __restrict__ Wo2,
                                                 const float* __restrict__ bo2,
                                                 float* __restrict__ out, int N) {
  __shared__ _Float16 sWs[16384];               // 32 KB, reused Wo1 then Wo2
  __shared__ _Float16 sA[4][16 * 136];
  const int tid = threadIdx.x;
  build_frags(sWs, Wo1, 4, tid);
  __syncthreads();

  const int lane = tid & 31;
  const int wave = tid >> 5;
  const int rowBase = (blockIdx.x * 4 + wave) * 16;
  const int m  = lane & 15;
  const int kb = (lane >> 4) << 3;
  const int mg = kb;
  const bool full = (rowBase + 16 <= N);

  int row = rowBase + m; if (row >= N) row = N - 1;
  const float* xr = acc + (long long)row * 128;
  v16h a[4];
  #pragma unroll
  for (int kk = 0; kk < 4; ++kk) a[kk] = cvt_frag_f32(xr + kk * 32 + kb);

  _Float16* myA = &sA[wave][0];
  #pragma unroll
  for (int t = 0; t < 8; ++t) {
    v8f c = {};
    #pragma unroll
    for (int kk = 0; kk < 4; ++kk) c = WMMA_F16(a[kk], bfrag(sWs, kk, t, lane), c);
    const int n = t * 16 + m;
    const float bias = bo1[n];
    _Float16* mp = myA + mg * 136 + n;
    #pragma unroll
    for (int rr = 0; rr < 8; ++rr) mp[rr * 136] = (_Float16)sspf(c[rr] + bias);
  }
  __syncthreads();                              // all waves done with Wo1 frags

  build_frags(sWs, Wo2, 4, tid);                // overwrite with Wo2 frags
  __syncthreads();

  v8f c2[8];
  v8f zero = {};
  #pragma unroll
  for (int t = 0; t < 8; ++t) c2[t] = zero;
  #pragma unroll
  for (int kk = 0; kk < 4; ++kk) {
    v16h a2 = lds_frag(myA + m * 136 + kk * 32 + kb);
    #pragma unroll
    for (int t = 0; t < 8; ++t) c2[t] = WMMA_F16(a2, bfrag(sWs, kk, t, lane), c2[t]);
  }

  #pragma unroll
  for (int t = 0; t < 8; ++t) {
    const int n = t * 16 + m;
    const float bias = bo2[n];
    float* op = out + (long long)(rowBase + mg) * 128 + n;
    if (full) {
      #pragma unroll
      for (int rr = 0; rr < 8; ++rr) op[rr * 128] = c2[t][rr] + bias;
    } else {
      #pragma unroll
      for (int rr = 0; rr < 8; ++rr)
        if (rowBase + mg + rr < N) op[rr * 128] = c2[t][rr] + bias;
    }
  }
}

// ---------------------------------------------------------------------------
extern "C" void kernel_launch(void* const* d_in, const int* in_sizes, int n_in,
                              void* d_out, int out_size, void* d_ws, size_t ws_size,
                              hipStream_t stream) {
  const float* x    = (const float*)d_in[0];
  const int*   pl   = (const int*)  d_in[1];   // [2, P]
  const float* fij  = (const float*)d_in[2];   // [P, 1, 32]
  const float* fcut = (const float*)d_in[3];   // [P, 1]
  const float* Win  = (const float*)d_in[4];
  const float* Wf1  = (const float*)d_in[5];
  const float* bf1  = (const float*)d_in[6];
  const float* Wf2  = (const float*)d_in[7];
  const float* bf2  = (const float*)d_in[8];
  const float* Wo1  = (const float*)d_in[9];
  const float* bo1  = (const float*)d_in[10];
  const float* Wo2  = (const float*)d_in[11];
  const float* bo2  = (const float*)d_in[12];
  float* out = (float*)d_out;

  const int       N = in_sizes[0] / 128;
  const long long P = in_sizes[1] / 2;

  float* h   = (float*)d_ws;                   // [N,128] f32 (L2-resident)
  float* acc = h + (long long)N * 128;         // [N,128] f32 accumulator

  hipMemsetAsync(acc, 0, (size_t)N * 128 * sizeof(float), stream);

  const int tilesN  = (N + 15) / 16;
  const int blocksN = (tilesN + 3) / 4;        // 4 waves (tiles) per block
  k_in_proj<<<blocksN, 128, 0, stream>>>(x, Win, h, N);

  const long long tilesP  = (P + 15) / 16;
  const int       blocksP = (int)((tilesP + 3) / 4);
  k_pairs<<<blocksP, 128, 0, stream>>>(fij, fcut, pl, Wf1, bf1, Wf2, bf2, h, acc, (int)P);

  k_out_mlp<<<blocksN, 128, 0, stream>>>(acc, Wo1, bo1, Wo2, bo2, out, N);
}
